// TensorProductScoreModel_9191230013568
// MI455X (gfx1250) — compile-verified, matrix operand order checked by reading, not run on hardware
//
#include <hip/hip_runtime.h>

// ---------------- problem constants ----------------
#define NEDGES   160000
#define NNODES   8000
#define IRR      56          // 2*16 + 6*4
#define K48      48          // EDGE_FEAT
#define WN       832         // W_NUMEL
#define OUTSZ    (NNODES*IRR)
#define WSTR     56          // padded LDS row stride (bf16 elems): 112B = 28 banks -> conflict-free b128

#define INV_SQRT3f 0.57735026918962576451f
#define INV_SQRT2f 0.70710678118654752440f
#define A_SCALf    0.22360679774997896964f   // 1/sqrt(20)
#define A_VECf     0.20412414523193150818f   // 1/sqrt(24)

// ---------------- WMMA types ----------------
typedef __attribute__((ext_vector_type(16))) __bf16        v16bf;
typedef __attribute__((ext_vector_type(8)))  float         v8f;
typedef __attribute__((ext_vector_type(4)))  unsigned int  u32x4;
typedef __attribute__((ext_vector_type(4)))  float         vf4;

union Frag { v16bf v; u32x4 q[2]; unsigned short e[16]; };

__device__ __forceinline__ unsigned short f2bf(float f) {
    unsigned u = __float_as_uint(f);
    unsigned r = u + 0x7FFFu + ((u >> 16) & 1u);   // round-to-nearest-even
    return (unsigned short)(r >> 16);
}

// ---------------- LDS layout (bytes) ----------------
// W2 bf16 [832][56]  : 93184
// W1 bf16 [48][56]   : 5376
// b2 f32  [832]      : 3328
// b1 f32  [48]       : 192
// 8 x per-wave region: staged multipliers (f32, transposed [u][16 edges]) + h (bf16, padded rows)
#define OFF_W2   0
#define OFF_W1   93184
#define OFF_B2   98560
#define OFF_B1   101888
#define OFF_WV   102080
#define PW_BYTES 9728
#define SMEM_BYTES (OFF_WV + 8*PW_BYTES)   // 179904

// per-wave offsets in floats
#define PX0E_S0 0
#define PX0E_T  256
#define PX0O_S0 512
#define PX0O_T  768
#define PDOTB   1024
#define PDOTC   1088
#define PX1O    1152   // [i][u][16] , s0 * x1o
#define PX1E    1344   // [i][u][16] , s0 * x1e
#define PCRE    1536   // [i][u][16] , cross(x1e,s1)*INV_SQRT2
#define PCRO    1728   // [i][u][16] , cross(x1o,s1)*INV_SQRT2
#define PS1T    1920   // [i][16]
#define PSRC    1968   // int [16]
#define PH_BYTES 7936  // h bf16 [16][WSTR] -> 1792 B

// ---------------- kernel 1: zero accumulators ----------------
__global__ void zero_kernel(float* __restrict__ out, float* __restrict__ cnt) {
    int i = blockIdx.x * 256 + threadIdx.x;
    if (i < OUTSZ) out[i] = 0.0f;
    else if (i < OUTSZ + NNODES) cnt[i - OUTSZ] = 0.0f;
}

// ---------------- kernel 2: fused MLP + tensor product + scatter ----------------
__global__ __launch_bounds__(256, 1)
void tp_conv_kernel(const float* __restrict__ node_attr,
                    const float* __restrict__ edge_attr,
                    const float* __restrict__ edge_sh,
                    const float* __restrict__ fc_w1,
                    const float* __restrict__ fc_b1,
                    const float* __restrict__ fc_w2,
                    const float* __restrict__ fc_b2,
                    const int*   __restrict__ edge_index,
                    float* __restrict__ out,
                    float* __restrict__ cnt)
{
    extern __shared__ char smem[];
    unsigned short* lw2 = (unsigned short*)(smem + OFF_W2);
    unsigned short* lw1 = (unsigned short*)(smem + OFF_W1);
    float* lb2 = (float*)(smem + OFF_B2);
    float* lb1 = (float*)(smem + OFF_B1);

    const int tid = threadIdx.x;

    // cooperative weight load: global [k][col] f32 -> LDS [col][k] bf16 (rows padded to WSTR)
    for (int i = tid; i < WN * K48; i += 256) {
        int col = i / K48, k = i - col * K48;
        lw2[col * WSTR + k] = f2bf(fc_w2[k * WN + col]);
    }
    for (int i = tid; i < K48 * K48; i += 256) {
        int col = i / K48, k = i - col * K48;
        lw1[col * WSTR + k] = f2bf(fc_w1[k * K48 + col]);
    }
    for (int i = tid; i < WN; i += 256) lb2[i] = fc_b2[i];
    if (tid < K48) lb1[tid] = fc_b1[tid];
    __syncthreads();

    const int wave = tid >> 5, lane = tid & 31;
    const int n = lane & 15, half = lane >> 4;
    const int ng = n >> 2;
    float* wf = (float*)(smem + OFF_WV + wave * PW_BYTES);
    int*   wsrc = (int*)(wf + PSRC);
    unsigned short* hst = (unsigned short*)((char*)wf + PH_BYTES);

    for (int it = 0; it < 5; ++it) {
        const int tile = it * 2000 + (int)blockIdx.x * 8 + wave;   // 10000 tiles total
        const int eb = tile * 16;

        // prefetch next iteration's edge_attr rows into cache (global_prefetch_b8)
        if (it < 4) {
            const int neb = (it + 1) * 2000 + (int)blockIdx.x * 8 + wave;
            __builtin_prefetch(edge_attr + (long)(neb * 16 + n) * K48 + half * 24, 0, 3);
        }

        // ---------- stage per-edge quantities (lanes 0..15, one edge each) ----------
        if (lane < 16) {
            int ge = eb + lane;
            wsrc[lane] = edge_index[ge];
            int d = edge_index[NEDGES + ge];
            float s0 = edge_sh[ge * 4 + 0];
            float sx = edge_sh[ge * 4 + 1];
            float sy = edge_sh[ge * 4 + 2];
            float sz = edge_sh[ge * 4 + 3];
            wf[PS1T + 0 * 16 + lane] = sx;
            wf[PS1T + 1 * 16 + lane] = sy;
            wf[PS1T + 2 * 16 + lane] = sz;
            const float* xa = node_attr + (long)d * IRR;
            #pragma unroll
            for (int u = 0; u < 16; ++u) {
                float v = xa[u];
                wf[PX0E_S0 + u * 16 + lane] = s0 * v;
                wf[PX0E_T  + u * 16 + lane] = v;
                float w = xa[40 + u];
                wf[PX0O_S0 + u * 16 + lane] = s0 * w;
                wf[PX0O_T  + u * 16 + lane] = w;
            }
            #pragma unroll
            for (int u = 0; u < 4; ++u) {
                float ax = xa[16 + u * 3 + 0], ay = xa[16 + u * 3 + 1], az = xa[16 + u * 3 + 2]; // x1o
                float bx = xa[28 + u * 3 + 0], by = xa[28 + u * 3 + 1], bz = xa[28 + u * 3 + 2]; // x1e
                wf[PDOTB + u * 16 + lane] = (ax * sx + ay * sy + az * sz) * INV_SQRT3f;
                wf[PDOTC + u * 16 + lane] = (bx * sx + by * sy + bz * sz) * INV_SQRT3f;
                wf[PX1O + (0 * 4 + u) * 16 + lane] = s0 * ax;
                wf[PX1O + (1 * 4 + u) * 16 + lane] = s0 * ay;
                wf[PX1O + (2 * 4 + u) * 16 + lane] = s0 * az;
                wf[PX1E + (0 * 4 + u) * 16 + lane] = s0 * bx;
                wf[PX1E + (1 * 4 + u) * 16 + lane] = s0 * by;
                wf[PX1E + (2 * 4 + u) * 16 + lane] = s0 * bz;
                wf[PCRE + (0 * 4 + u) * 16 + lane] = (by * sz - bz * sy) * INV_SQRT2f;
                wf[PCRE + (1 * 4 + u) * 16 + lane] = (bz * sx - bx * sz) * INV_SQRT2f;
                wf[PCRE + (2 * 4 + u) * 16 + lane] = (bx * sy - by * sx) * INV_SQRT2f;
                wf[PCRO + (0 * 4 + u) * 16 + lane] = (ay * sz - az * sy) * INV_SQRT2f;
                wf[PCRO + (1 * 4 + u) * 16 + lane] = (az * sx - ax * sz) * INV_SQRT2f;
                wf[PCRO + (2 * 4 + u) * 16 + lane] = (ax * sy - ay * sx) * INV_SQRT2f;
            }
        }
        __syncthreads();

        // ---------- GEMM1 A fragments: edge_attr tile (16x48) as bf16, K split 32+16 ----------
        Frag a0, a1;
        {
            const float* er = edge_attr + (long)(eb + n) * K48;
            vf4 p0 = *(const vf4*)(er + half * 8);
            vf4 p1 = *(const vf4*)(er + half * 8 + 4);
            vf4 p2 = *(const vf4*)(er + 16 + half * 8);
            vf4 p3 = *(const vf4*)(er + 16 + half * 8 + 4);
            vf4 p4 = *(const vf4*)(er + 32 + half * 8);
            vf4 p5 = *(const vf4*)(er + 32 + half * 8 + 4);
            #pragma unroll
            for (int j = 0; j < 4; ++j) {
                a0.e[j]      = f2bf(p0[j]);  a0.e[4 + j]  = f2bf(p1[j]);
                a0.e[8 + j]  = f2bf(p2[j]);  a0.e[12 + j] = f2bf(p3[j]);
                a1.e[j]      = f2bf(p4[j]);  a1.e[4 + j]  = f2bf(p5[j]);
                a1.e[8 + j]  = 0;            a1.e[12 + j] = 0;
            }
        }

        // ---------- GEMM1: h = relu(ea @ W1 + b1) -> LDS (bf16, [edge][WSTR]) ----------
        #pragma unroll
        for (int nt = 0; nt < 3; ++nt) {
            int col = nt * 16 + n;
            const unsigned short* br = lw1 + col * WSTR;
            Frag b0, b1;
            b0.q[0] = *(const u32x4*)(br + half * 8);
            b0.q[1] = *(const u32x4*)(br + 16 + half * 8);
            b1.q[0] = *(const u32x4*)(br + 32 + half * 8);
            b1.q[1] = (u32x4){0u, 0u, 0u, 0u};
            v8f c = {0, 0, 0, 0, 0, 0, 0, 0};
            c = __builtin_amdgcn_wmma_f32_16x16x32_bf16(false, a0.v, false, b0.v, (short)0, c, false, false);
            c = __builtin_amdgcn_wmma_f32_16x16x32_bf16(false, a1.v, false, b1.v, (short)0, c, false, false);
            float bias = lb1[col];
            #pragma unroll
            for (int r = 0; r < 8; ++r) {
                float hv = fmaxf(c[r] + bias, 0.0f);
                hst[(r + 8 * half) * WSTR + col] = f2bf(hv);
            }
        }
        __syncthreads();

        // ---------- A fragments of h ----------
        Frag ha0, ha1;
        {
            const unsigned short* hr = hst + n * WSTR;
            ha0.q[0] = *(const u32x4*)(hr + half * 8);
            ha0.q[1] = *(const u32x4*)(hr + 16 + half * 8);
            ha1.q[0] = *(const u32x4*)(hr + 32 + half * 8);
            ha1.q[1] = (u32x4){0u, 0u, 0u, 0u};
        }

        // ---------- GEMM2 tile-by-tile, fused tensor-product accumulation ----------
        v8f accA  = {0,0,0,0,0,0,0,0}, accD  = {0,0,0,0,0,0,0,0};
        v8f accP1 = {0,0,0,0,0,0,0,0}, accP2 = {0,0,0,0,0,0,0,0};
        v8f aO1o0 = {0,0,0,0,0,0,0,0}, aO1o1 = {0,0,0,0,0,0,0,0}, aO1o2 = {0,0,0,0,0,0,0,0};
        v8f aO1e0 = {0,0,0,0,0,0,0,0}, aO1e1 = {0,0,0,0,0,0,0,0}, aO1e2 = {0,0,0,0,0,0,0,0};

        auto wtile = [&](int t) -> v8f {
            int col = t * 16 + n;
            const unsigned short* br = lw2 + col * WSTR;
            Frag b0, b1;
            b0.q[0] = *(const u32x4*)(br + half * 8);
            b0.q[1] = *(const u32x4*)(br + 16 + half * 8);
            b1.q[0] = *(const u32x4*)(br + 32 + half * 8);
            b1.q[1] = (u32x4){0u, 0u, 0u, 0u};
            v8f c = {0, 0, 0, 0, 0, 0, 0, 0};
            c = __builtin_amdgcn_wmma_f32_16x16x32_bf16(false, ha0.v, false, b0.v, (short)0, c, false, false);
            c = __builtin_amdgcn_wmma_f32_16x16x32_bf16(false, ha1.v, false, b1.v, (short)0, c, false, false);
            float bias = lb2[col];
            #pragma unroll
            for (int r = 0; r < 8; ++r) c[r] += bias;
            return c;
        };
        auto fmacc = [&](v8f& acc, const float* base, v8f wv) {
            vf4 m0 = *(const vf4*)(base + half * 8);
            vf4 m1 = *(const vf4*)(base + half * 8 + 4);
            #pragma unroll
            for (int r = 0; r < 4; ++r) { acc[r] += m0[r] * wv[r]; acc[4 + r] += m1[r] * wv[4 + r]; }
        };

        for (int t = 0; t < 16; ++t) { v8f wv = wtile(t);      fmacc(accA, wf + PX0E_S0 + t * 16, wv); }      // x0e*wA0*s0
        for (int t = 0; t < 4;  ++t) { v8f wv = wtile(16 + t); fmacc(accA, wf + PDOTB + t * 16, wv); }        // dot_b*wB0
        for (int t = 0; t < 4;  ++t) { v8f wv = wtile(20 + t); fmacc(accP1, wf + PX0E_T + (t * 4 + ng) * 16, wv); } // x0e*wA1
        { v8f wv = wtile(24);
          fmacc(aO1o0, wf + PX1O + (0 * 4 + ng) * 16, wv);
          fmacc(aO1o1, wf + PX1O + (1 * 4 + ng) * 16, wv);
          fmacc(aO1o2, wf + PX1O + (2 * 4 + ng) * 16, wv); }                                                  // s0*x1o*wB1
        { v8f wv = wtile(25);
          fmacc(aO1o0, wf + PCRE + (0 * 4 + ng) * 16, wv);
          fmacc(aO1o1, wf + PCRE + (1 * 4 + ng) * 16, wv);
          fmacc(aO1o2, wf + PCRE + (2 * 4 + ng) * 16, wv); }                                                  // cr_e*wC1
        { v8f wv = wtile(26);
          fmacc(aO1e0, wf + PCRO + (0 * 4 + ng) * 16, wv);
          fmacc(aO1e1, wf + PCRO + (1 * 4 + ng) * 16, wv);
          fmacc(aO1e2, wf + PCRO + (2 * 4 + ng) * 16, wv); }                                                  // cr_o*wB1e
        { v8f wv = wtile(27);
          fmacc(aO1e0, wf + PX1E + (0 * 4 + ng) * 16, wv);
          fmacc(aO1e1, wf + PX1E + (1 * 4 + ng) * 16, wv);
          fmacc(aO1e2, wf + PX1E + (2 * 4 + ng) * 16, wv); }                                                  // s0*x1e*wC1e
        for (int t = 0; t < 4;  ++t) { v8f wv = wtile(28 + t); fmacc(accP2, wf + PX0O_T + (t * 4 + ng) * 16, wv); } // x0o*wD1e
        for (int t = 0; t < 4;  ++t) { v8f wv = wtile(32 + t); fmacc(accD, wf + PDOTC + t * 16, wv); }        // dot_c*wC0o
        for (int t = 0; t < 16; ++t) { v8f wv = wtile(36 + t); fmacc(accD, wf + PX0O_S0 + t * 16, wv); }      // s0*x0o*wD0o

        // ---------- writeout: scatter-add by src ----------
        int srcv[8];
        #pragma unroll
        for (int r = 0; r < 8; ++r) srcv[r] = wsrc[r + 8 * half];

        #pragma unroll
        for (int r = 0; r < 8; ++r) {
            long base = (long)srcv[r] * IRR;
            atomicAdd(out + base + n,      A_SCALf * accA[r]);   // o0e
            atomicAdd(out + base + 40 + n, A_SCALf * accD[r]);   // o0o
        }
        // fold partial sums over u (lanes v, v+4, v+8, v+12 within each half)
        #pragma unroll
        for (int r = 0; r < 8; ++r) {
            accP1[r] += __shfl_xor(accP1[r], 4); accP1[r] += __shfl_xor(accP1[r], 8);
            accP2[r] += __shfl_xor(accP2[r], 4); accP2[r] += __shfl_xor(accP2[r], 8);
            aO1o0[r] += __shfl_xor(aO1o0[r], 4); aO1o0[r] += __shfl_xor(aO1o0[r], 8);
            aO1o1[r] += __shfl_xor(aO1o1[r], 4); aO1o1[r] += __shfl_xor(aO1o1[r], 8);
            aO1o2[r] += __shfl_xor(aO1o2[r], 4); aO1o2[r] += __shfl_xor(aO1o2[r], 8);
            aO1e0[r] += __shfl_xor(aO1e0[r], 4); aO1e0[r] += __shfl_xor(aO1e0[r], 8);
            aO1e1[r] += __shfl_xor(aO1e1[r], 4); aO1e1[r] += __shfl_xor(aO1e1[r], 8);
            aO1e2[r] += __shfl_xor(aO1e2[r], 4); aO1e2[r] += __shfl_xor(aO1e2[r], 8);
        }
        if (n < 4) {   // lane's column == output v index
            #pragma unroll
            for (int r = 0; r < 8; ++r) {
                long base = (long)srcv[r] * IRR;
                float sx = wf[PS1T + 0 * 16 + 8 * half + r];
                float sy = wf[PS1T + 1 * 16 + 8 * half + r];
                float sz = wf[PS1T + 2 * 16 + 8 * half + r];
                atomicAdd(out + base + 16 + n * 3 + 0, A_VECf * (aO1o0[r] + accP1[r] * sx));
                atomicAdd(out + base + 16 + n * 3 + 1, A_VECf * (aO1o1[r] + accP1[r] * sy));
                atomicAdd(out + base + 16 + n * 3 + 2, A_VECf * (aO1o2[r] + accP1[r] * sz));
                atomicAdd(out + base + 28 + n * 3 + 0, A_VECf * (aO1e0[r] + accP2[r] * sx));
                atomicAdd(out + base + 28 + n * 3 + 1, A_VECf * (aO1e1[r] + accP2[r] * sy));
                atomicAdd(out + base + 28 + n * 3 + 2, A_VECf * (aO1e2[r] + accP2[r] * sz));
            }
        }
        if (lane < 16) atomicAdd(cnt + wsrc[lane], 1.0f);
        __syncthreads();   // protect staging area before next iteration
    }
}

// ---------------- kernel 3: divide by segment counts ----------------
__global__ void finalize_kernel(float* __restrict__ out, const float* __restrict__ cnt) {
    int i = blockIdx.x * 256 + threadIdx.x;
    if (i < OUTSZ) {
        float c = cnt[i / IRR];
        out[i] = out[i] / fmaxf(c, 1.0f);
    }
}

extern "C" void kernel_launch(void* const* d_in, const int* in_sizes, int n_in,
                              void* d_out, int out_size, void* d_ws, size_t ws_size,
                              hipStream_t stream) {
    const float* node_attr  = (const float*)d_in[0];
    const float* edge_attr  = (const float*)d_in[1];
    const float* edge_sh    = (const float*)d_in[2];
    const float* fc_w1      = (const float*)d_in[3];
    const float* fc_b1      = (const float*)d_in[4];
    const float* fc_w2      = (const float*)d_in[5];
    const float* fc_b2      = (const float*)d_in[6];
    const int*   edge_index = (const int*)d_in[7];
    float* out = (float*)d_out;
    float* cnt = (float*)d_ws;

    zero_kernel<<<(OUTSZ + NNODES + 255) / 256, 256, 0, stream>>>(out, cnt);
    // 250 blocks * 8 waves * 5 iters = 10000 tiles of 16 edges = 160000 edges
    tp_conv_kernel<<<250, 256, SMEM_BYTES, stream>>>(node_attr, edge_attr, edge_sh,
                                                     fc_w1, fc_b1, fc_w2, fc_b2,
                                                     edge_index, out, cnt);
    finalize_kernel<<<(OUTSZ + 255) / 256, 256, 0, stream>>>(out, cnt);
}
